// MultiHeadAttention_38259568673287
// MI455X (gfx1250) — matrix operand order repaired and checked by reading, not compiled, not run on hardware
//
#include <hip/hip_runtime.h>
#include <hip/hip_bf16.h>
#include <stdint.h>

// ---------------- problem constants ----------------
#define D_MODEL 1024
#define NHEADS  16
#define HDIM    64
#define BATCH   2
#define SEQ     2048
#define MROWS   (BATCH * SEQ)   // 4096 total rows
#define SROW    (SEQ + 9)       // LDS score-row stride (odd -> bank-conflict free)

typedef __bf16 bf16_t;
typedef __attribute__((ext_vector_type(16))) __bf16 v16bf;
typedef __attribute__((ext_vector_type(8)))  __bf16 v8bf;
typedef __attribute__((ext_vector_type(8)))  float  v8f;
typedef __attribute__((ext_vector_type(8)))  float  f32x8;

// ---------------------------------------------------------------------------
// WMMA fragment layouts (wave32, V_WMMA_F32_16X16X32_BF16, ISA 7.12.2)
//  A (16x32): lanes 0-15 row M=lane, K=k0..k0+15 (contiguous order used here);
//             lanes 16-31 row M=lane-16, K=k0+16..k0+31.
//  B (32x16): lane holds column N=lane&15; same k order as A (not the ISA table): lanes 0-15 K=kk..kk+15,
//             lanes 16-31 K=kk+16..kk+31 -> one contiguous 16-elem load from a
//             K-contiguous ("transposed") source row.
//  C/D (16x16 f32): lane L, VGPR r -> M = r + (L>=16 ? 8 : 0), N = L&15.
// ---------------------------------------------------------------------------

__device__ __forceinline__ v16bf load_a_bf16(const bf16_t* __restrict__ row, int basek) {
    v8bf c0 = *(const v8bf*)(row + basek);
    v8bf c1 = *(const v8bf*)(row + basek + 8);
    v16bf a;
#pragma unroll
    for (int i = 0; i < 8; ++i) { a[i] = c0[i]; a[i + 8] = c1[i]; }
    return a;
}

__device__ __forceinline__ v8f wmma_bf16(v16bf a, v16bf b, v8f c) {
    return __builtin_amdgcn_wmma_f32_16x16x32_bf16(false, a, false, b, (short)0, c,
                                                   false, false);
}

// ---------------------------------------------------------------------------
// One-time fp32 -> bf16 conversion (activations reused 64x, weights 256x by
// the GEMMs: converting once keeps all v_cvt out of the WMMA hot loops).
// n must be a multiple of 8.
// ---------------------------------------------------------------------------
__global__ void cvt_f32_bf16(const float* __restrict__ src, bf16_t* __restrict__ dst,
                             int n) {
    int i = (blockIdx.x * blockDim.x + threadIdx.x) * 8;
    if (i >= n) return;
    f32x8 v = *(const f32x8*)(src + i);
    v8bf o;
#pragma unroll
    for (int e = 0; e < 8; ++e) o[e] = (__bf16)v[e];
    *(v8bf*)(dst + i) = o;
}

// ---------------------------------------------------------------------------
// Projection GEMM (pure bf16):  C[m,n] = sum_k A[m,k] * W[n,k] + bias[n]
//   A: [MROWS x D_MODEL] bf16 row-major; W: [D_MODEL x D_MODEL] bf16 row-major.
// Each wave computes a 16x64 strip: one A fragment feeds 4 WMMAs per k-step
// (4x arithmetic intensity vs one-tile waves; all operands are L2-resident).
//   mode 0: store bf16, per-head row-major   dst[((b*H+h)*S+s)*HDIM + d]  (Q,K)
//   mode 1: store bf16, per-head TRANSPOSED  dst[((b*H+h)*HDIM+d)*S + s]  (V)
//   mode 2: store fp32 plain [MROWS x D_MODEL]                            (out)
// ---------------------------------------------------------------------------
__global__ void gemm_bf16(const bf16_t* __restrict__ A,
                          const bf16_t* __restrict__ W,
                          const float* __restrict__ bias,
                          void* __restrict__ dst, int mode) {
    const int lane = threadIdx.x & 31;
    const int wave = threadIdx.x >> 5;
    const int job  = blockIdx.x * (blockDim.x >> 5) + wave;
    const int NTG  = D_MODEL / 64;           // 16 column groups of 64
    const int mt   = job / NTG;
    const int ng   = job % NTG;
    if (mt >= MROWS / 16) return;            // wave-uniform; never taken

    const int laneN  = lane & 15;
    const int laneHi = lane >> 4;
    const int m      = mt * 16 + laneN;
    const int nbase  = ng * 64;

    const bf16_t* Arow = A + (size_t)m * D_MODEL;
    const bf16_t* Wr0  = W + (size_t)(nbase +  0 + laneN) * D_MODEL;
    const bf16_t* Wr1  = W + (size_t)(nbase + 16 + laneN) * D_MODEL;
    const bf16_t* Wr2  = W + (size_t)(nbase + 32 + laneN) * D_MODEL;
    const bf16_t* Wr3  = W + (size_t)(nbase + 48 + laneN) * D_MODEL;

    v8f acc0 = {}, acc1 = {}, acc2 = {}, acc3 = {};
#pragma unroll 2
    for (int k0 = 0; k0 < D_MODEL; k0 += 32) {
        v16bf a  = load_a_bf16(Arow, k0 + laneHi * 16);
        int   kb = k0 + laneHi * 16;
        v16bf b0 = *(const v16bf*)(Wr0 + kb);
        v16bf b1 = *(const v16bf*)(Wr1 + kb);
        v16bf b2 = *(const v16bf*)(Wr2 + kb);
        v16bf b3 = *(const v16bf*)(Wr3 + kb);
        acc0 = wmma_bf16(a, b0, acc0);
        acc1 = wmma_bf16(a, b1, acc1);
        acc2 = wmma_bf16(a, b2, acc2);
        acc3 = wmma_bf16(a, b3, acc3);
    }

#pragma unroll
    for (int t = 0; t < 4; ++t) {
        v8f acc = (t == 0) ? acc0 : (t == 1) ? acc1 : (t == 2) ? acc2 : acc3;
        const int   n  = nbase + t * 16 + laneN;
        const float bn = bias[n];
        if (mode == 2) {
            float* out = (float*)dst;
#pragma unroll
            for (int r = 0; r < 8; ++r) {
                int mr = mt * 16 + r + laneHi * 8;
                out[(size_t)mr * D_MODEL + n] = acc[r] + bn;
            }
        } else {
            bf16_t* out = (bf16_t*)dst;
            const int h = n / HDIM, d = n % HDIM;
#pragma unroll
            for (int r = 0; r < 8; ++r) {
                int   mr = mt * 16 + r + laneHi * 8;
                int   bb = mr / SEQ, s = mr % SEQ;
                float v  = acc[r] + bn;
                if (mode == 0)
                    out[(((size_t)bb * NHEADS + h) * SEQ + s) * HDIM + d] = (bf16_t)v;
                else
                    out[(((size_t)bb * NHEADS + h) * HDIM + d) * SEQ + s] = (bf16_t)v;
            }
        }
    }
}

// ---------------------------------------------------------------------------
// Fused attention: one workgroup (4 waves / 128 thr) per 16 query rows of one
// (b,h). Score strip [16 x SEQ] lives in LDS (~131 KB of the 320 KB/WGP).
//   phase 1: QK^T via WMMA (each wave covers 512 key columns), scale + mask
//   phase 2: cooperative softmax in LDS, coalesced fp32 store of `w`
//   phase 3: ctx[16x64] = w @ V via WMMA (each wave one 16-col tile),
//            bf16 store into concat layout for the output projection.
// Q/K/V bf16 (24 MB) and mask (16 MB) are L2-resident (192 MB L2), so HBM
// traffic ~= the mandatory 536 MB write of `w` (~23 us at 23.3 TB/s).
// ---------------------------------------------------------------------------
__global__ void attn_fused(const bf16_t* __restrict__ qb,
                           const bf16_t* __restrict__ kb,
                           const bf16_t* __restrict__ vb,   // transposed [bh][d][s]
                           const float* __restrict__ mask,  // [SEQ x SEQ]
                           float* __restrict__ wout,        // [B,H,S,S]
                           bf16_t* __restrict__ cb) {       // [B*S x D_MODEL] concat
    extern __shared__ float smem[];
    float* sc  = smem;               // [16][SROW]
    float* red = smem + 16 * SROW;   // [16*8] partials, [16] max @128, [16] 1/sum @144

    const int tid    = threadIdx.x;
    const int lane   = tid & 31;
    const int wave   = tid >> 5;
    const int laneN  = lane & 15;
    const int laneHi = lane >> 4;

    const int qblk = blockIdx.x % (SEQ / 16);
    const int bh   = blockIdx.x / (SEQ / 16);
    const int h    = bh % NHEADS;
    const int b    = bh / NHEADS;
    const int i0   = qblk * 16;

    const bf16_t* Q = qb + (size_t)bh * SEQ * HDIM;
    const bf16_t* K = kb + (size_t)bh * SEQ * HDIM;
    const bf16_t* V = vb + (size_t)bh * HDIM * SEQ;

    // ---- phase 1: scores = Q K^T / sqrt(dh) + mask * -1e9 -> LDS ----
    const float inv_sqrt = 0.125f;   // 1/sqrt(64)
    const bf16_t* Qrow = Q + (size_t)(i0 + laneN) * HDIM;
    const v16bf a0 = load_a_bf16(Qrow, 0  + laneHi * 16);   // d = 0..31
    const v16bf a1 = load_a_bf16(Qrow, 32 + laneHi * 16);   // d = 32..63

    for (int j0 = wave * (SEQ / 4); j0 < (wave + 1) * (SEQ / 4); j0 += 16) {
        const bf16_t* Krow = K + (size_t)(j0 + laneN) * HDIM;  // column j0+laneN
        v16bf b0 = *(const v16bf*)(Krow + laneHi * 16);        // d = 0..31
        v16bf b1 = *(const v16bf*)(Krow + 32 + laneHi * 16);   // d = 32..63
        v8f acc = {};
        acc = wmma_bf16(a0, b0, acc);
        acc = wmma_bf16(a1, b1, acc);
        const int j = j0 + laneN;
#pragma unroll
        for (int r = 0; r < 8; ++r) {
            int   mloc = r + laneHi * 8;
            int   i    = i0 + mloc;
            float sv   = acc[r] * inv_sqrt + mask[(size_t)i * SEQ + j] * (-1e9f);
            sc[mloc * SROW + j] = sv;
        }
    }
    __syncthreads();

    // ---- phase 2: softmax over each of the 16 rows (8 threads per row) ----
    const int row = tid >> 3;   // 0..15
    const int ch  = tid & 7;    // 0..7, 256-wide chunks
    float* rowp = sc + row * SROW;

    float mx = -3.4e38f;
    for (int j = ch * 256; j < ch * 256 + 256; ++j) mx = fmaxf(mx, rowp[j]);
    red[row * 8 + ch] = mx;
    __syncthreads();
    if (ch == 0) {
        float m2 = red[row * 8];
#pragma unroll
        for (int t = 1; t < 8; ++t) m2 = fmaxf(m2, red[row * 8 + t]);
        red[128 + row] = m2;
    }
    __syncthreads();

    const float rmax = red[128 + row];
    float sm = 0.f;
    for (int j = ch * 256; j < ch * 256 + 256; ++j) {
        float e = __expf(rowp[j] - rmax);
        rowp[j] = e;
        sm += e;
    }
    red[row * 8 + ch] = sm;
    __syncthreads();
    if (ch == 0) {
        float s2 = red[row * 8];
#pragma unroll
        for (int t = 1; t < 8; ++t) s2 += red[row * 8 + t];
        red[144 + row] = 1.0f / s2;
    }
    __syncthreads();

    // normalize in LDS + coalesced store of w (the only large HBM write)
    float* wbase = wout + ((size_t)bh * SEQ + i0) * SEQ;
    for (int idx = tid; idx < 16 * SEQ; idx += 128) {
        int   mloc = idx >> 11;         // / SEQ
        int   j    = idx & (SEQ - 1);
        float v    = sc[mloc * SROW + j] * red[144 + mloc];
        sc[mloc * SROW + j] = v;
        wbase[(size_t)mloc * SEQ + j] = v;
    }
    __syncthreads();

    // ---- phase 3: ctx[16 x 16] per wave = w[16 x S] @ V[S x 16] ----
    const int d0 = wave * 16;
    v8f cacc = {};
    const int mloc = laneN;
    for (int kk = 0; kk < SEQ; kk += 32) {
        const int basek = kk + laneHi * 16;
        v16bf aw;
#pragma unroll
        for (int e = 0; e < 8; ++e) {
            aw[e]     = (__bf16)sc[mloc * SROW + basek + e];
            aw[e + 8] = (__bf16)sc[mloc * SROW + basek + 8 + e];
        }
        const bf16_t* Vrow = V + (size_t)(d0 + laneN) * SEQ + kk + laneHi * 16;
        v16bf bw = *(const v16bf*)Vrow;
        cacc = wmma_bf16(aw, bw, cacc);
    }
    const int d   = d0 + laneN;
    const int col = h * HDIM + d;
#pragma unroll
    for (int r = 0; r < 8; ++r) {
        int srow = i0 + r + laneHi * 8;
        cb[((size_t)b * SEQ + srow) * D_MODEL + col] = (bf16_t)cacc[r];
    }
}

// ---------------------------------------------------------------------------
extern "C" void kernel_launch(void* const* d_in, const int* in_sizes, int n_in,
                              void* d_out, int out_size, void* d_ws, size_t ws_size,
                              hipStream_t stream) {
    (void)in_sizes; (void)n_in; (void)out_size; (void)ws_size;

    const float* q    = (const float*)d_in[0];
    const float* k    = (const float*)d_in[1];
    const float* v    = (const float*)d_in[2];
    const float* mask = (const float*)d_in[3];
    const float* Wk   = (const float*)d_in[4];
    const float* bk   = (const float*)d_in[5];
    const float* Wv   = (const float*)d_in[6];
    const float* bv   = (const float*)d_in[7];
    const float* Wo   = (const float*)d_in[8];
    const float* bo   = (const float*)d_in[9];

    float* out  = (float*)d_out;                                   // [B,S,D]
    float* wout = out + (size_t)BATCH * SEQ * D_MODEL;             // [B,H,S,S]

    // workspace layout (bf16 buffers)
    char*        ws  = (char*)d_ws;
    const size_t ACT = (size_t)MROWS * D_MODEL * sizeof(bf16_t);     // 8 MB
    const size_t WGT = (size_t)D_MODEL * D_MODEL * sizeof(bf16_t);   // 2 MB
    bf16_t* qb  = (bf16_t*)(ws + 0 * ACT);   // projected Q (per-head rows)
    bf16_t* kbf = (bf16_t*)(ws + 1 * ACT);   // projected K (per-head rows)
    bf16_t* vbf = (bf16_t*)(ws + 2 * ACT);   // projected V (per-head transposed)
    bf16_t* cb  = (bf16_t*)(ws + 3 * ACT);   // ctx concat [B*S x D]
    bf16_t* qf  = (bf16_t*)(ws + 4 * ACT);   // bf16(query)
    bf16_t* kf  = (bf16_t*)(ws + 5 * ACT);   // bf16(key)
    bf16_t* vf  = (bf16_t*)(ws + 6 * ACT);   // bf16(value)
    bf16_t* Wkb = (bf16_t*)(ws + 7 * ACT + 0 * WGT);
    bf16_t* Wvb = (bf16_t*)(ws + 7 * ACT + 1 * WGT);
    bf16_t* Wob = (bf16_t*)(ws + 7 * ACT + 2 * WGT);

    // one-time f32 -> bf16 conversions (keeps v_cvt out of WMMA loops)
    const int NA = MROWS * D_MODEL;      // 4 Mi elements
    const int NW = D_MODEL * D_MODEL;    // 1 Mi elements
    cvt_f32_bf16<<<dim3(NA / 8 / 256), dim3(256), 0, stream>>>(q,  qf,  NA);
    cvt_f32_bf16<<<dim3(NA / 8 / 256), dim3(256), 0, stream>>>(k,  kf,  NA);
    cvt_f32_bf16<<<dim3(NA / 8 / 256), dim3(256), 0, stream>>>(v,  vf,  NA);
    cvt_f32_bf16<<<dim3(NW / 8 / 256), dim3(256), 0, stream>>>(Wk, Wkb, NW);
    cvt_f32_bf16<<<dim3(NW / 8 / 256), dim3(256), 0, stream>>>(Wv, Wvb, NW);
    cvt_f32_bf16<<<dim3(NW / 8 / 256), dim3(256), 0, stream>>>(Wo, Wob, NW);

    // projections (NOTE: reference bug kept — Q projected with Wk/bk)
    const int  jobs = (MROWS / 16) * (D_MODEL / 64);   // 4096 wave-jobs
    const dim3 blk(256);                               // 8 waves/block
    const dim3 grd(jobs / 8);                          // 512 blocks
    gemm_bf16<<<grd, blk, 0, stream>>>(qf, Wkb, bk, qb,  0);
    gemm_bf16<<<grd, blk, 0, stream>>>(kf, Wkb, bk, kbf, 0);
    gemm_bf16<<<grd, blk, 0, stream>>>(vf, Wvb, bv, vbf, 1);

    const size_t smem = (size_t)(16 * SROW + 160) * sizeof(float); // ~132 KB
    attn_fused<<<dim3(BATCH * NHEADS * (SEQ / 16)), dim3(128), smem, stream>>>(
        qb, kbf, vbf, mask, wout, cb);

    gemm_bf16<<<grd, blk, 0, stream>>>(cb, Wob, bo, out, 2);
}
